// BinaryDense_77223511982716
// MI455X (gfx1250) — 8-bit matrix operand order repaired and checked by reading, compiled, not run on hardware
//
#include <hip/hip_runtime.h>
#include <hip/hip_bf16.h>

typedef __attribute__((ext_vector_type(16))) int      v16i;
typedef __attribute__((ext_vector_type(8)))  int      v8i;
typedef __attribute__((ext_vector_type(4)))  int      v4i;
typedef __attribute__((ext_vector_type(4)))  unsigned v4u;
typedef __attribute__((ext_vector_type(8)))  float    v8f;
typedef __attribute__((ext_vector_type(4)))  float    f32x4;

#define IN_FEATURES 256
#define UNITS       256
#define BATCH       65536
#define B_DWORDS    16384

__device__ __forceinline__ unsigned sign_fp8(float x) {
    unsigned bits = __float_as_uint(x);
    unsigned mag  = (x != 0.0f) ? 0x38u : 0u;
    return mag | ((bits >> 24) & 0x80u);
}

__device__ __forceinline__ unsigned pack4(f32x4 f) {
    return sign_fp8(f.x) | (sign_fp8(f.y) << 8) | (sign_fp8(f.z) << 16) | (sign_fp8(f.w) << 24);
}

__global__ void __launch_bounds__(256)
pack_w_fp8_kernel(const float* __restrict__ w, unsigned* __restrict__ bpack) {
    const int tid  = blockIdx.x * blockDim.x + threadIdx.x;
    const int tt   = tid >> 9;
    const int lane = (tid >> 4) & 31;
    const int v    = tid & 15;

    const int kh    = tt >> 4;
    const int ntile = tt & 15;
    const int n     = ntile * 16 + (lane & 15);
    const int g     = v >> 2;
    const int j     = v & 3;
    const int k     = kh * 128 + g * 32 + ((lane >> 4) * 16) + j * 4;

    unsigned d = 0;
#pragma unroll
    for (int b = 0; b < 4; ++b) {
        d |= sign_fp8(w[(size_t)(k + b) * UNITS + n]) << (8 * b);
    }
    bpack[((size_t)(tt * 4 + g) * 32 + lane) * 4 + j] = d;
}

__global__ void __launch_bounds__(256)
binary_dense_wmma_kernel(const float* __restrict__ x,
                         const unsigned* __restrict__ bpack,
                         float* __restrict__ out) {
    __shared__ unsigned lds_b[B_DWORDS];

    const int lane = threadIdx.x & 31;
    const int wave = threadIdx.x >> 5;

#if __has_builtin(__builtin_amdgcn_tensor_load_to_lds)
    if (wave == 0) {
        unsigned long long ga = (unsigned long long)(const void*)bpack;
        v4u g0;
        g0[0] = 1u;
        g0[1] = 0u;
        g0[2] = (unsigned)(ga & 0xFFFFFFFFull);
        g0[3] = (unsigned)((ga >> 32) & 0x1FFFFFFull) | (2u << 30);
        v8i g1;
        g1[0] = (int)(2u << 16);
        g1[1] = (int)(16384u << 16);
        g1[2] = (int)(1u << 16);
        g1[3] = (int)(16384u << 16);
        g1[4] = 1;
        g1[5] = 16384;
        g1[6] = 0;
        g1[7] = 0;
        v4i z4 = {0, 0, 0, 0};
#if defined(__clang_major__) && (__clang_major__ >= 23)
        v8i z8 = {0, 0, 0, 0, 0, 0, 0, 0};
        __builtin_amdgcn_tensor_load_to_lds(g0, g1, z4, z4, z8, 0);
#else
        __builtin_amdgcn_tensor_load_to_lds(g0, g1, z4, z4, 0);
#endif
        __builtin_amdgcn_s_wait_tensorcnt((short)0);
    }
    __syncthreads();
#else
    for (int i = threadIdx.x; i < B_DWORDS; i += 256) lds_b[i] = bpack[i];
    __syncthreads();
#endif

    asm volatile("" : : "v"(&lds_b[0]) : "memory");
    const v4i* __restrict__ lb = (const v4i*)lds_b;

    const int strip    = blockIdx.x * 8 + wave;
    const int row_base = strip * 16;

    const int m = row_base + (lane & 15);
    const float* __restrict__ xrow = x + (size_t)m * IN_FEATURES;
    const int khi = (lane >> 4) * 16;

    v16i a0, a1;
#pragma unroll
    for (int v = 0; v < 16; ++v) {
        const int v8_ = v & 7;
        const int kk  = (v >> 2) * 32 + khi + (v & 3) * 4;
        a0[v] = (int)pack4(*(const f32x4*)(xrow + kk));
        a1[v] = (int)pack4(*(const f32x4*)(xrow + 128 + kk));
    }

#pragma unroll
    for (int t = 0; t < 16; ++t) {
        v16i b0, b1;
#pragma unroll
        for (int g = 0; g < 4; ++g) {
            const v4i c0 = lb[(t * 4 + g) * 32 + lane];
            const v4i c1 = lb[((16 + t) * 4 + g) * 32 + lane];
#pragma unroll
            for (int j = 0; j < 4; ++j) {
                b0[4 * g + j] = c0[j];
                b1[4 * g + j] = c1[j];
            }
        }

        v8f acc = {};
        acc = __builtin_amdgcn_wmma_f32_16x16x128_fp8_fp8(a0, b0, (short)0, acc, false, false);
        acc = __builtin_amdgcn_wmma_f32_16x16x128_fp8_fp8(a1, b1, (short)0, acc, false, false);

        const int ncol = t * 16 + (lane & 15);
        const int moff = (lane >> 4) * 8;
#pragma unroll
        for (int r = 0; r < 8; ++r) {
            out[(size_t)(row_base + moff + r) * UNITS + ncol] = acc[r];
        }
    }
}

extern "C" void kernel_launch(void* const* d_in, const int* in_sizes, int n_in,
                              void* d_out, int out_size, void* d_ws, size_t ws_size,
                              hipStream_t stream) {
    const float* x  = (const float*)d_in[0];
    const float* w  = (const float*)d_in[1];
    float* out      = (float*)d_out;
    unsigned* bpack = (unsigned*)d_ws;

    pack_w_fp8_kernel<<<64, 256, 0, stream>>>(w, bpack);
    binary_dense_wmma_kernel<<<BATCH / 128, 256, 0, stream>>>(x, bpack, out);
}
